// TransformerBlock_13709535609485
// MI455X (gfx1250) — compile-verified, matrix operand order checked by reading, not run on hardware
//
#include <hip/hip_runtime.h>

typedef unsigned short u16;
typedef __attribute__((ext_vector_type(16))) __bf16 v16bf;
typedef __attribute__((ext_vector_type(8)))  __bf16 v8bf;
typedef __attribute__((ext_vector_type(8)))  float  v8f;
typedef __attribute__((ext_vector_type(4)))  unsigned u32x4;
typedef __attribute__((ext_vector_type(8)))  unsigned u32x8;

#define B_  4
#define S_  2048
#define D_  1024
#define H_  16
#define HD_ 64
#define M_  (B_ * S_)

#define TK        32                 // K-step (bf16 WMMA K)
#define LDP       40                 // padded LDS tile pitch in elements
#define TILE_U16  (128 * LDP)        // u16 elements per LDS tile buffer
#define TILE_BYTES (TILE_U16 * 2)    // 10240 B

// ---------------------------------------------------------------- helpers

__device__ __forceinline__ u16 f2bf(float f) {
  unsigned u = __float_as_uint(f);
  u += 0x7fffu + ((u >> 16) & 1u);      // round-to-nearest-even
  return (u16)(u >> 16);
}

// 16-bit A-matrix 16x32 fragment (ISA 7.12.2):
//   lanes 0-15 : row = lane,    VGPR0-3 = K[0..7],   VGPR4-7 = K[16..23]
//   lanes16-31 : row = lane-16, VGPR0-3 = K[8..15],  VGPR4-7 = K[24..31]
// => per lane: two contiguous 16B loads at column koff and koff+16.
// B fragments use the same pattern applied to B^T (lane = N index).
__device__ __forceinline__ v16bf load_frag(const u16* __restrict__ base,
                                           int row0, int pitch, int kbase) {
  const int lane = threadIdx.x & 31;
  const int row  = row0 + (lane & 15);
  const int koff = (lane >> 4) << 3;                 // 0 or 8
  const u16* p = base + (size_t)row * pitch + kbase + koff;
  v8bf lo = *(const v8bf*)(p);
  v8bf hi = *(const v8bf*)(p + 16);
  return __builtin_shufflevector(lo, hi, 0,1,2,3,4,5,6,7,8,9,10,11,12,13,14,15);
}

__device__ __forceinline__ v8f wmma_bf16(v16bf a, v16bf b, v8f c) {
  return __builtin_amdgcn_wmma_f32_16x16x32_bf16(false, a, false, b,
                                                 (short)0, c, false, false);
}

// Tensor Data Mover: 2D tile load, 32 cols x 128 rows of bf16, memory row
// pitch = Kdim elements, packed into LDS with 4 pad DWORDs per 16-DWORD row
// (=> LDS pitch LDP=40 elements).  D# per ISA 08_async_tensor 8.3/8.4.
__device__ __forceinline__ void tdm_load_tile(unsigned lds_off, const u16* gsrc,
                                              unsigned rows_total, unsigned Kdim) {
  const unsigned long long ga = (unsigned long long)(const void*)gsrc;
  u32x4 g0;
  g0[0] = 1u;                                   // count=1 valid user descriptor
  g0[1] = lds_off;                              // lds_addr (bytes)
  g0[2] = (unsigned)ga;                         // global_addr[31:0]
  g0[3] = (unsigned)(ga >> 32) | (2u << 30);    // global_addr[56:32] | type=2
  u32x8 g1;
  g1[0] = (1u << 16)                            // data_size = 2 bytes
        | (1u << 20)                            // pad_enable
        | (3u << 22)                            // pad_interval: 16 DWORDs
        | (3u << 25);                           // pad_amount: 4 DWORDs
  g1[1] = (Kdim & 0xFFFFu) << 16;               // tensor_dim0[15:0]
  g1[2] = (Kdim >> 16) | ((rows_total & 0xFFFFu) << 16);   // dim0 hi | dim1 lo
  g1[3] = (rows_total >> 16) | ((unsigned)TK << 16);       // dim1 hi | tile_dim0
  g1[4] = 128u;                                 // tile_dim1=128, tile_dim2=0
  g1[5] = Kdim;                                 // tensor_dim0_stride[31:0]
  g1[6] = 0u;                                   // stride0 hi | stride1 lo
  g1[7] = 0u;
  asm volatile("tensor_load_to_lds %0, %1" :: "s"(g0), "s"(g1) : "memory");
}

// ---------------------------------------------------------------- convert

__global__ __launch_bounds__(256) void cvt_k(const float* __restrict__ in,
                                             u16* __restrict__ out, int n) {
  const int i = (blockIdx.x * 256 + threadIdx.x) * 4;
  if (i < n) {
    const float4 f = *(const float4*)(in + i);
    out[i + 0] = f2bf(f.x); out[i + 1] = f2bf(f.y);
    out[i + 2] = f2bf(f.z); out[i + 3] = f2bf(f.w);
  }
}

__global__ void cvt_wT_k(const float* __restrict__ W, u16* __restrict__ WT) {
  const int k = blockIdx.x * 16 + threadIdx.x;
  const int n = blockIdx.y * 16 + threadIdx.y;
  WT[(size_t)n * D_ + k] = f2bf(W[(size_t)k * D_ + n]);
}

// Pack bool mask into bits: one u32 per 32 keys (wave32 ballot).
__global__ __launch_bounds__(256) void pack_mask_k(
    const unsigned char* __restrict__ m, unsigned* __restrict__ out) {
  const int i = blockIdx.x * 256 + threadIdx.x;
  const unsigned w = (unsigned)__ballot(m[i] != 0);
  if ((threadIdx.x & 31) == 0) out[i >> 5] = w;
}

// ---------------------------------------------------------------- GEMM
// C[M,N] = A[M,K] * B[K,N] + bias.  A bf16 row-major, B supplied as B^T bf16
// row-major.  Block = 8 waves -> 128x128 tile; wave = 32x64 (2x4 frags).
// A/B 128x32 tiles staged to LDS by the Tensor Data Mover, double-buffered.
// MODE 0: bf16 row-major   MODE 1: relu + bf16   MODE 2: fp32
// MODE 3: bf16 per-head transposed V layout  Vt[b][h][hd][s]

template <int MODE>
__global__ __launch_bounds__(256) void gemm_k(
    const u16* __restrict__ A, const u16* __restrict__ Bt,
    const float* __restrict__ bias, u16* __restrict__ outb,
    float* __restrict__ outf, int Mdim, int Ndim, int Kdim) {
  // [bufA0][bufA1][bufB0][bufB1]; single shared object => LDS offset 0.
  __shared__ __align__(16) u16 smem[4 * TILE_U16];
  const int lane = threadIdx.x & 31;
  const int wid  = threadIdx.x >> 5;
  const int mblk = blockIdx.x * 128;
  const int nblk = blockIdx.y * 128;
  const int wm   = (wid >> 1) * 32;
  const int wn   = (wid & 1) * 64;
  const int nk   = Kdim / TK;

  if (wid == 0) {
    tdm_load_tile(0u,             A  + (size_t)mblk * Kdim, (unsigned)Mdim, (unsigned)Kdim);
    tdm_load_tile(2u * TILE_BYTES, Bt + (size_t)nblk * Kdim, (unsigned)Ndim, (unsigned)Kdim);
    __builtin_amdgcn_s_wait_tensorcnt(0);
  }
  __syncthreads();

  v8f acc[2][4];
  v8f z = {};
#pragma unroll
  for (int i = 0; i < 2; ++i)
#pragma unroll
    for (int j = 0; j < 4; ++j) acc[i][j] = z;

  for (int kb = 0; kb < nk; ++kb) {
    const int cur = kb & 1;
    if (wid == 0 && kb + 1 < nk) {            // prefetch next K-slab via TDM
      const int k2 = (kb + 1) * TK;
      tdm_load_tile((unsigned)((1 - cur) * TILE_BYTES),
                    A + (size_t)mblk * Kdim + k2, (unsigned)Mdim, (unsigned)Kdim);
      tdm_load_tile((unsigned)((2 + (1 - cur)) * TILE_BYTES),
                    Bt + (size_t)nblk * Kdim + k2, (unsigned)Ndim, (unsigned)Kdim);
    }
    const u16* ldsA = &smem[cur * TILE_U16];
    const u16* ldsB = &smem[(2 + cur) * TILE_U16];
    v16bf a0 = load_frag(ldsA, wm,      LDP, 0);
    v16bf a1 = load_frag(ldsA, wm + 16, LDP, 0);
#pragma unroll
    for (int ni = 0; ni < 4; ++ni) {
      v16bf b = load_frag(ldsB, wn + ni * 16, LDP, 0);
      acc[0][ni] = wmma_bf16(a0, b, acc[0][ni]);
      acc[1][ni] = wmma_bf16(a1, b, acc[1][ni]);
    }
    if (wid == 0 && kb + 1 < nk) __builtin_amdgcn_s_wait_tensorcnt(0);
    __syncthreads();
  }

  // C layout: VGPR r -> row r (lanes 0-15) / row r+8 (lanes 16-31), col=lane&15
  const int rgrp = (lane >> 4) << 3;
  const int ncol = lane & 15;
#pragma unroll
  for (int mi = 0; mi < 2; ++mi) {
#pragma unroll
    for (int ni = 0; ni < 4; ++ni) {
      const int   n  = nblk + wn + ni * 16 + ncol;
      const float bv = bias[n];
#pragma unroll
      for (int r = 0; r < 8; ++r) {
        const int m = mblk + wm + mi * 16 + rgrp + r;
        float v = acc[mi][ni][r] + bv;
        if (MODE == 1) v = fmaxf(v, 0.0f);
        if (MODE == 2) {
          outf[(size_t)m * Ndim + n] = v;
        } else if (MODE == 3) {
          const int bb = m >> 11, ss = m & (S_ - 1);
          const int hh = n >> 6,  hd = n & (HD_ - 1);
          outb[((size_t)((bb * H_ + hh) * HD_ + hd)) * S_ + ss] = f2bf(v);
        } else {
          outb[(size_t)m * Ndim + n] = f2bf(v);
        }
      }
    }
  }
}

// ---------------------------------------------------------------- attention
// One wave per (b, h, 16-row q tile), flash-style online softmax over
// 32-key blocks.  Q/K read from row-major [B*S, D] bf16, V from transposed
// Vt[b][h][hd][s] bf16.  Mask consumed as packed bits (1 b32 load / block).

__global__ __launch_bounds__(256) void attn_k(
    const u16* __restrict__ Qb, const u16* __restrict__ Kb,
    const u16* __restrict__ Vt, const unsigned* __restrict__ mbits,
    u16* __restrict__ Ob) {
  __shared__ __align__(16) u16 plds[8 * 16 * 40];   // per-wave P transpose buf
  const int lane = threadIdx.x & 31;
  const int wid  = threadIdx.x >> 5;
  const int g    = blockIdx.x * 8 + wid;
  const int b    = g >> 11;           // / (H_ * S_/16)
  const int rem  = g & 2047;
  const int h    = rem >> 7;
  const int q0   = (rem & 127) << 4;

  const u16* Qbase = Qb + (size_t)b * S_ * D_ + h * HD_;
  const u16* Kbase = Kb + (size_t)b * S_ * D_ + h * HD_;
  const u16* Vbase = Vt + (size_t)(b * H_ + h) * HD_ * S_;
  u16* myp = plds + wid * (16 * 40);

  const v16bf qa0 = load_frag(Qbase, q0, D_, 0);
  const v16bf qa1 = load_frag(Qbase, q0, D_, 32);

  const int rgrp = (lane >> 4) << 3;
  const int ncol = lane & 15;
  const float NEG_INF = -__builtin_inff();
  // lane loads the packed-mask word for "its" row; lanes 0-15 cover rows
  // 0..7 (x2), lanes 16-31 cover rows 8..15 (x2); __shfl(.,r,8) broadcasts.
  const size_t mrowidx = ((size_t)b * S_ + q0 + rgrp + (lane & 7)) * (S_ >> 5);

  float mi[8], li[8];
  v8f o[4];
  v8f z = {};
#pragma unroll
  for (int r = 0; r < 8; ++r) { mi[r] = -1e30f; li[r] = 0.0f; }
#pragma unroll
  for (int t = 0; t < 4; ++t) o[t] = z;

  for (int j = 0; j < S_; j += 32) {
    if (j + 32 < S_)                        // global_prefetch next K rows
      __builtin_prefetch(Kbase + (size_t)(j + 32 + lane) * D_, 0, 1);

    v8f s0 = z, s1 = z;
    s0 = wmma_bf16(qa0, load_frag(Kbase, j,      D_, 0),  s0);
    s0 = wmma_bf16(qa1, load_frag(Kbase, j,      D_, 32), s0);
    s1 = wmma_bf16(qa0, load_frag(Kbase, j + 16, D_, 0),  s1);
    s1 = wmma_bf16(qa1, load_frag(Kbase, j + 16, D_, 32), s1);

    const unsigned wrow = mbits[mrowidx + (j >> 5)];
#pragma unroll
    for (int r = 0; r < 8; ++r) {
      const unsigned wr = __shfl(wrow, r, 8);   // this row's 32 mask bits
      float v0 = s0[r] * 0.125f;                // 1/sqrt(64)
      float v1 = s1[r] * 0.125f;
      if (!((wr >> ncol) & 1u))        v0 = NEG_INF;
      if (!((wr >> (16 + ncol)) & 1u)) v1 = NEG_INF;

      float rm = fmaxf(v0, v1);                 // row max over 16 lanes
      rm = fmaxf(rm, __shfl_xor(rm, 1, 16));
      rm = fmaxf(rm, __shfl_xor(rm, 2, 16));
      rm = fmaxf(rm, __shfl_xor(rm, 4, 16));
      rm = fmaxf(rm, __shfl_xor(rm, 8, 16));
      const float mnew  = fmaxf(mi[r], rm);
      const float alpha = __expf(mi[r] - mnew);
      mi[r] = mnew;
      const float p0 = __expf(v0 - mnew);
      const float p1 = __expf(v1 - mnew);
      float rs = p0 + p1;
      rs += __shfl_xor(rs, 1, 16);
      rs += __shfl_xor(rs, 2, 16);
      rs += __shfl_xor(rs, 4, 16);
      rs += __shfl_xor(rs, 8, 16);
      li[r] = li[r] * alpha + rs;
      o[0][r] *= alpha; o[1][r] *= alpha; o[2][r] *= alpha; o[3][r] *= alpha;

      // P from C layout -> LDS (row = rgrp+r, col = block col)
      myp[(rgrp + r) * 40 + ncol]      = f2bf(p0);
      myp[(rgrp + r) * 40 + 16 + ncol] = f2bf(p1);
    }

    asm volatile("s_wait_dscnt 0" ::: "memory");  // cross-lane DS RAW

    const v16bf pa = load_frag(myp, 0, 40, 0);    // re-read as A fragment
#pragma unroll
    for (int t = 0; t < 4; ++t)
      o[t] = wmma_bf16(pa, load_frag(Vbase, t * 16, S_, j), o[t]);
  }

#pragma unroll
  for (int t = 0; t < 4; ++t) {
#pragma unroll
    for (int r = 0; r < 8; ++r) {
      const int m = q0 + rgrp + r;
      Ob[((size_t)b * S_ + m) * D_ + h * HD_ + t * 16 + ncol] =
          f2bf(o[t][r] / li[r]);
    }
  }
}

// ---------------------------------------------------------------- layernorm

__device__ __forceinline__ float block_sum(float v, float* red) {
#pragma unroll
  for (int off = 16; off > 0; off >>= 1) v += __shfl_xor(v, off, 32);
  if ((threadIdx.x & 31) == 0) red[threadIdx.x >> 5] = v;
  __syncthreads();
  float t = (threadIdx.x < 8) ? red[threadIdx.x] : 0.0f;
  t += __shfl_xor(t, 1, 32);
  t += __shfl_xor(t, 2, 32);
  t += __shfl_xor(t, 4, 32);
  if (threadIdx.x == 0) red[8] = t;
  __syncthreads();
  const float r = red[8];
  __syncthreads();
  return r;
}

__global__ __launch_bounds__(256) void ln_k(
    const float* __restrict__ a, const float* __restrict__ res,
    const float* __restrict__ scale, const float* __restrict__ bias,
    float* __restrict__ outf, u16* __restrict__ outb) {
  __shared__ float red[9];
  const size_t row = blockIdx.x;
  const int tid = threadIdx.x;
  float v[4];
  float s = 0.0f;
#pragma unroll
  for (int k = 0; k < 4; ++k) {
    const int i = tid + k * 256;
    v[k] = a[row * D_ + i] + res[row * D_ + i];
    s += v[k];
  }
  const float mu = block_sum(s, red) * (1.0f / D_);
  float q = 0.0f;
#pragma unroll
  for (int k = 0; k < 4; ++k) { const float d = v[k] - mu; q += d * d; }
  const float rstd = rsqrtf(block_sum(q, red) * (1.0f / D_) + 1e-6f);
#pragma unroll
  for (int k = 0; k < 4; ++k) {
    const int i = tid + k * 256;
    const float o = (v[k] - mu) * rstd * scale[i] + bias[i];
    if (outf) outf[row * D_ + i] = o;
    if (outb) outb[row * D_ + i] = f2bf(o);
  }
}

// ---------------------------------------------------------------- launch

extern "C" void kernel_launch(void* const* d_in, const int* in_sizes, int n_in,
                              void* d_out, int out_size, void* d_ws,
                              size_t ws_size, hipStream_t stream) {
  const float* x    = (const float*)d_in[0];
  const unsigned char* mask = (const unsigned char*)d_in[1];
  const float* Wq = (const float*)d_in[3];  const float* bq = (const float*)d_in[4];
  const float* Wk = (const float*)d_in[5];  const float* bk = (const float*)d_in[6];
  const float* Wv = (const float*)d_in[7];  const float* bv = (const float*)d_in[8];
  const float* Wo = (const float*)d_in[9];  const float* bo = (const float*)d_in[10];
  const float* W1 = (const float*)d_in[11]; const float* b1 = (const float*)d_in[12];
  const float* W2 = (const float*)d_in[13]; const float* b2 = (const float*)d_in[14];
  const float* ln1s = (const float*)d_in[15]; const float* ln1b = (const float*)d_in[16];
  const float* ln2s = (const float*)d_in[17]; const float* ln2b = (const float*)d_in[18];

  char* ws = (char*)d_ws;
  size_t off = 0;
  auto alloc = [&](size_t bytes) -> void* {
    void* p = ws + off;
    off += (bytes + 255) & ~(size_t)255;
    return p;
  };
  u16* xb    = (u16*)alloc((size_t)M_ * D_ * 2);
  u16* WqT   = (u16*)alloc((size_t)D_ * D_ * 2);
  u16* WkT   = (u16*)alloc((size_t)D_ * D_ * 2);
  u16* WvT   = (u16*)alloc((size_t)D_ * D_ * 2);
  u16* WoT   = (u16*)alloc((size_t)D_ * D_ * 2);
  u16* W1T   = (u16*)alloc((size_t)D_ * D_ * 2);
  u16* W2T   = (u16*)alloc((size_t)D_ * D_ * 2);
  unsigned* mbits = (unsigned*)alloc((size_t)B_ * S_ * (S_ / 32) * 4);
  u16* Qb    = (u16*)alloc((size_t)M_ * D_ * 2);
  u16* Kb    = (u16*)alloc((size_t)M_ * D_ * 2);
  u16* Vt    = (u16*)alloc((size_t)M_ * D_ * 2);
  u16* valsb = (u16*)alloc((size_t)M_ * D_ * 2);
  float* f32a = (float*)alloc((size_t)M_ * D_ * 4);
  float* hbuf = (float*)alloc((size_t)M_ * D_ * 4);
  // dead-buffer reuse (stream-ordered, no overlap within a launch):
  u16*   hb   = Qb;            // post-LN1 bf16 activations
  u16*   ff1  = Kb;            // relu(h@W1+b1) bf16
  float* f32b = (float*)Vt;    // FFN2 fp32 out spans Vt+valsb (32 MiB)

  cvt_k<<<(M_ * D_) / 1024, 256, 0, stream>>>(x, xb, M_ * D_);
  {
    dim3 tb(16, 16), tg(D_ / 16, D_ / 16);
    cvt_wT_k<<<tg, tb, 0, stream>>>(Wq, WqT);
    cvt_wT_k<<<tg, tb, 0, stream>>>(Wk, WkT);
    cvt_wT_k<<<tg, tb, 0, stream>>>(Wv, WvT);
    cvt_wT_k<<<tg, tb, 0, stream>>>(Wo, WoT);
    cvt_wT_k<<<tg, tb, 0, stream>>>(W1, W1T);
    cvt_wT_k<<<tg, tb, 0, stream>>>(W2, W2T);
  }
  pack_mask_k<<<(B_ * S_ * S_) / 256, 256, 0, stream>>>(mask, mbits);

  dim3 gg(M_ / 128, D_ / 128);
  gemm_k<0><<<gg, 256, 0, stream>>>(xb, WqT, bq, Qb, nullptr, M_, D_, D_);
  gemm_k<0><<<gg, 256, 0, stream>>>(xb, WkT, bk, Kb, nullptr, M_, D_, D_);
  gemm_k<3><<<gg, 256, 0, stream>>>(xb, WvT, bv, Vt, nullptr, M_, D_, D_);

  attn_k<<<(B_ * H_ * (S_ / 16)) / 8, 256, 0, stream>>>(Qb, Kb, Vt, mbits, valsb);

  gemm_k<2><<<gg, 256, 0, stream>>>(valsb, WoT, bo, nullptr, f32a, M_, D_, D_);
  ln_k<<<M_, 256, 0, stream>>>(x, f32a, ln1s, ln1b, hbuf, hb);
  gemm_k<1><<<gg, 256, 0, stream>>>(hb, W1T, b1, ff1, nullptr, M_, D_, D_);
  gemm_k<2><<<gg, 256, 0, stream>>>(ff1, W2T, b2, nullptr, f32b, M_, D_, D_);
  ln_k<<<M_, 256, 0, stream>>>(hbuf, f32b, ln2s, ln2b, (float*)d_out, nullptr);

  (void)in_sizes; (void)n_in; (void)out_size; (void)ws_size;
}
